// EncoderBlock_60541859004700
// MI455X (gfx1250) — compile-verified, matrix operand order checked by reading, not run on hardware
//
#include <hip/hip_runtime.h>

typedef unsigned short u16;
typedef __attribute__((ext_vector_type(16))) __bf16 bf16x16;
typedef __attribute__((ext_vector_type(8)))  float   floatx8;

// ---------------- helpers ----------------

static __device__ __forceinline__ u16 f2bf(float f) {
  union { float f; unsigned u; } a; a.f = f;
  unsigned u = a.u;
  u += 0x7FFFu + ((u >> 16) & 1u);   // round-to-nearest-even
  return (u16)(u >> 16);
}

static __device__ __forceinline__ floatx8 zero8() {
  floatx8 z;
  #pragma unroll
  for (int i = 0; i < 8; ++i) z[i] = 0.0f;
  return z;
}

static __device__ __forceinline__ floatx8 wmma_bf16(bf16x16 a, bf16x16 b, floatx8 c) {
  // D = A(16x32) * B(32x16) + C, f32 accumulate
  return __builtin_amdgcn_wmma_f32_16x16x32_bf16(false, a, false, b, (short)0, c, false, false);
}

// Async DMA: global -> LDS, 16 bytes per lane, tracked by ASYNCcnt (CDNA5).
static __device__ __forceinline__ void async_load_b128(unsigned lds_byte_off, const void* gaddr) {
  asm volatile("global_load_async_to_lds_b128 %0, %1, off"
               :: "v"(lds_byte_off), "v"((unsigned long long)(uintptr_t)gaddr)
               : "memory");
}
static __device__ __forceinline__ void wait_async0() {
  asm volatile("s_wait_asynccnt 0x0" ::: "memory");
}
static __device__ __forceinline__ unsigned lds_off(const void* p) {
  return (unsigned)(uintptr_t)p;   // low 32 bits of generic pointer = LDS address
}

// Load a 16-bit WMMA fragment from an LDS tile stored row-major [row][k]
// (A: row = M index; B: tile stored transposed so row = N index).
// Lane layout (wave32): lanes 0-15 hold K 0..7 & 16..23, lanes 16-31 hold K 8..15 & 24..31.
static __device__ __forceinline__ bf16x16 load_frag(const u16* tile, int row, int stride, int kbase) {
  const int half = (threadIdx.x >> 4) & 1;
  const u16* p = tile + row * stride + kbase + half * 8;
  union { uint4 q[2]; bf16x16 v; } f;
  f.q[0] = *(const uint4*)(p);
  f.q[1] = *(const uint4*)(p + 16);
  return f.v;
}

// One BK=64 compute stage: 16 ds_load_b128 then 8 chained WMMAs (acc stays tied in place).
static __device__ __forceinline__ void compute_stage(
    const u16* Ac, const u16* Bc, int wm, int wn, int lrow, floatx8 (&acc)[2][2])
{
  bf16x16 a00 = load_frag(Ac, wm * 32 + 0  + lrow, 72, 0);
  bf16x16 a10 = load_frag(Ac, wm * 32 + 16 + lrow, 72, 0);
  bf16x16 b00 = load_frag(Bc, wn * 32 + 0  + lrow, 72, 0);
  bf16x16 b10 = load_frag(Bc, wn * 32 + 16 + lrow, 72, 0);
  bf16x16 a01 = load_frag(Ac, wm * 32 + 0  + lrow, 72, 32);
  bf16x16 a11 = load_frag(Ac, wm * 32 + 16 + lrow, 72, 32);
  bf16x16 b01 = load_frag(Bc, wn * 32 + 0  + lrow, 72, 32);
  bf16x16 b11 = load_frag(Bc, wn * 32 + 16 + lrow, 72, 32);
  acc[0][0] = wmma_bf16(a01, b01, wmma_bf16(a00, b00, acc[0][0]));
  acc[0][1] = wmma_bf16(a01, b11, wmma_bf16(a00, b10, acc[0][1]));
  acc[1][0] = wmma_bf16(a11, b01, wmma_bf16(a10, b00, acc[1][0]));
  acc[1][1] = wmma_bf16(a11, b11, wmma_bf16(a10, b10, acc[1][1]));
}

// ---------------- generic 128x64 GEMM core, BK = 64, async double-buffered ----------------
// 256 threads = 8 waves; wave w: rows 32*(w&3).., cols 32*(w>>2)..; 2x2 WMMA tiles per wave.
// A : row-major [M x K] bf16 (pre-offset to block row 0), lda in elems.
// Bt: TRANSPOSED [N x K] bf16 (pre-offset to block col 0), ldb in elems (= K).
// K must be a multiple of 128 (even number of BK=64 stages).

static __device__ __forceinline__ void gemm_core_128x64(
    const u16* __restrict__ A, int lda,
    const u16* __restrict__ Bt, int ldb, int K,
    u16* As0, u16* As1, u16* Bs0, u16* Bs1, floatx8 (&acc)[2][2])
{
  const int tid  = threadIdx.x;
  const int lane = tid & 31;
  const int wave = tid >> 5;
  const int wm   = wave & 3;
  const int wn   = wave >> 2;
  const int lrow = lane & 15;
  const int arow = tid >> 1, ac0 = (tid & 1) * 32;
  const int brow = tid >> 2, bc0 = (tid & 3) * 16;

  const u16* gA = A + (size_t)arow * lda + ac0;
  const u16* gB = Bt + (size_t)brow * ldb + bc0;

  // issue tile k0 into (AsBuf, BsBuf): A 128x64 (4x16B/thread), B 64x64 (2x16B/thread)
  auto issue_tile = [&](int k0, u16* AsBuf, u16* BsBuf) {
    unsigned la = lds_off(AsBuf + arow * 72 + ac0);
    const u16* pa = gA + k0;
    async_load_b128(la +  0, pa +  0);
    async_load_b128(la + 16, pa +  8);
    async_load_b128(la + 32, pa + 16);
    async_load_b128(la + 48, pa + 24);
    unsigned lb = lds_off(BsBuf + brow * 72 + bc0);
    const u16* pb = gB + k0;
    async_load_b128(lb +  0, pb +  0);
    async_load_b128(lb + 16, pb +  8);
  };

  const int nstage = K >> 6;   // even
  issue_tile(0, As0, Bs0);

  for (int t = 0; t < nstage; t += 2) {
    wait_async0();        // tile t complete (this wave's DMA)
    __syncthreads();      // tile t visible to all waves; buf1 free (read two stages ago)
    if (t + 1 < nstage) issue_tile((t + 1) << 6, As1, Bs1);
    compute_stage(As0, Bs0, wm, wn, lrow, acc);

    wait_async0();        // tile t+1 complete
    __syncthreads();      // tile t+1 visible; buf0 free (all waves done with compute above)
    if (t + 2 < nstage) issue_tile((t + 2) << 6, As0, Bs0);
    compute_stage(As1, Bs1, wm, wn, lrow, acc);
  }
}

// ---------------- kernels ----------------

__global__ void pack_bf16_kernel(const float* __restrict__ in, u16* __restrict__ out, int n) {
  int i = blockIdx.x * blockDim.x + threadIdx.x;
  int stride = gridDim.x * blockDim.x;
  for (; i < n; i += stride) out[i] = f2bf(in[i]);
}

// Pack + transpose: in [1024 x N] row-major f32 -> out [N x 1024] bf16. One matrix per blockIdx.z.
__global__ void packT_bf16_kernel(const float* __restrict__ in, u16* __restrict__ out, int N) {
  const float* src = in + (size_t)blockIdx.z * 1024 * N;
  u16* dst = out + (size_t)blockIdx.z * 1024 * N;
  int i = blockIdx.x * blockDim.x + threadIdx.x;   // over N*1024, k fastest (coalesced writes)
  int total = N * 1024;
  int stride = gridDim.x * blockDim.x;
  for (; i < total; i += stride) {
    int n = i >> 10, k = i & 1023;
    dst[i] = f2bf(src[(size_t)k * N + n]);
  }
}

// Fused per-head QKV projection. grid = (32 m-tiles, 16 heads, 3 types). M=4096,N=64,K=1024.
// Q/K layout: [h][r][64] with r = b*1024 + s.  V layout (pre-transposed): [h][b][d][s].
__global__ __launch_bounds__(256) void qkv_kernel(
    const u16* __restrict__ xb,
    const u16* __restrict__ Wqt, const u16* __restrict__ Wkt, const u16* __restrict__ Wvt,
    const float* __restrict__ bq, const float* __restrict__ bk, const float* __restrict__ bv,
    u16* __restrict__ Q, u16* __restrict__ Ko, u16* __restrict__ Vt)
{
  __shared__ u16 As0[128 * 72];
  __shared__ u16 As1[128 * 72];
  __shared__ u16 Bs0[64 * 72];
  __shared__ u16 Bs1[64 * 72];
  const int mt   = blockIdx.x;
  const int h    = blockIdx.y;
  const int type = blockIdx.z;
  const u16*   W    = (type == 0) ? Wqt : (type == 1) ? Wkt : Wvt;
  const float* bias = (type == 0) ? bq  : (type == 1) ? bk  : bv;
  u16*         Out  = ((type == 0) ? Q : (type == 1) ? Ko : Vt) + (size_t)h * 4096 * 64;

  floatx8 acc[2][2];
  acc[0][0] = zero8(); acc[0][1] = zero8(); acc[1][0] = zero8(); acc[1][1] = zero8();

  gemm_core_128x64(xb + (size_t)mt * 128 * 1024, 1024,
                   W + (size_t)h * 64 * 1024, 1024, 1024, As0, As1, Bs0, Bs1, acc);

  const int lane = threadIdx.x & 31, wave = threadIdx.x >> 5;
  const int wm = wave & 3, wn = wave >> 2, half = lane >> 4, lc = lane & 15;
  #pragma unroll
  for (int mtt = 0; mtt < 2; ++mtt)
    #pragma unroll
    for (int ntt = 0; ntt < 2; ++ntt)
      #pragma unroll
      for (int v = 0; v < 8; ++v) {
        int r = mt * 128 + wm * 32 + mtt * 16 + v + half * 8;   // global row (b*1024+s)
        int c = wn * 32 + ntt * 16 + lc;                        // head dim 0..63
        float val = acc[mtt][ntt][v] + bias[h * 64 + c];
        if (type == 2) {
          int b = r >> 10, s = r & 1023;
          Out[((size_t)b * 64 + c) * 1024 + s] = f2bf(val);     // V transposed [d][s]
        } else {
          Out[(size_t)r * 64 + c] = f2bf(val);
        }
      }
}

// Flash-style fused attention. grid = (16 q-tiles, 16 heads, 4 batch). 256 threads.
__global__ __launch_bounds__(256) void attn_kernel(
    const u16* __restrict__ Q, const u16* __restrict__ Kk,
    const u16* __restrict__ Vt, u16* __restrict__ ctx)
{
  __shared__ u16   Qs[64 * 72];
  __shared__ u16   Ks[64 * 72];
  __shared__ u16   Vs[64 * 72];   // [d][key]
  __shared__ float Ss[64 * 68];
  __shared__ u16   Ps[64 * 72];
  __shared__ float rowm[64], rowl[64], rowalpha[64], part[256];

  const int qt = blockIdx.x, h = blockIdx.y, b = blockIdx.z;
  const size_t hb = ((size_t)h * 4 + b) * (1024 * 64);
  const u16* Qh = Q  + hb;
  const u16* Kh = Kk + hb;
  const u16* Vh = Vt + hb;   // [d][s] layout

  const int tid = threadIdx.x, lane = tid & 31, wave = tid >> 5;
  const int wm = wave & 3, wn2 = (wave >> 2) * 2, half = lane >> 4, lc = lane & 15;
  const int ldr = tid >> 2, ldc0 = (tid & 3) * 16;

  { // Q tile (64x64) -> Qs[m][k]
    const uint4* s = (const uint4*)(Qh + (size_t)(qt * 64 + ldr) * 64 + ldc0);
    uint4* d = (uint4*)(Qs + ldr * 72 + ldc0);
    d[0] = s[0]; d[1] = s[1];
  }
  if (tid < 64) { rowm[tid] = -1e30f; rowl[tid] = 0.0f; }
  floatx8 o0 = zero8(), o1 = zero8();
  __syncthreads();

  for (int t = 0; t < 16; ++t) {
    { // K tile: Ks[n=key][k=d]  (B layout == K row-major)
      const uint4* s = (const uint4*)(Kh + (size_t)(t * 64 + ldr) * 64 + ldc0);
      uint4* d = (uint4*)(Ks + ldr * 72 + ldc0);
      d[0] = s[0]; d[1] = s[1];
    }
    { // V tile: Vs[n=d][k=key]  (global already transposed)
      const uint4* s = (const uint4*)(Vh + (size_t)ldr * 1024 + t * 64 + ldc0);
      uint4* d = (uint4*)(Vs + ldr * 72 + ldc0);
      d[0] = s[0]; d[1] = s[1];
    }
    __syncthreads();

    // scores tile S = Q * K^T * scale
    floatx8 s0 = zero8(), s1 = zero8();
    {
      bf16x16 aq0 = load_frag(Qs, wm * 16 + lc, 72, 0);
      bf16x16 aq1 = load_frag(Qs, wm * 16 + lc, 72, 32);
      bf16x16 bk00 = load_frag(Ks, (wn2 + 0) * 16 + lc, 72, 0);
      bf16x16 bk01 = load_frag(Ks, (wn2 + 0) * 16 + lc, 72, 32);
      bf16x16 bk10 = load_frag(Ks, (wn2 + 1) * 16 + lc, 72, 0);
      bf16x16 bk11 = load_frag(Ks, (wn2 + 1) * 16 + lc, 72, 32);
      s0 = wmma_bf16(aq1, bk01, wmma_bf16(aq0, bk00, s0));
      s1 = wmma_bf16(aq1, bk11, wmma_bf16(aq0, bk10, s1));
    }
    #pragma unroll
    for (int v = 0; v < 8; ++v) {
      int r = wm * 16 + v + half * 8;
      Ss[r * 68 + (wn2 + 0) * 16 + lc] = s0[v] * 0.125f;
      Ss[r * 68 + (wn2 + 1) * 16 + lc] = s1[v] * 0.125f;
    }
    __syncthreads();

    { // per-row partial max (4 threads / row)
      int r = tid >> 2, seg = tid & 3;
      float m = -1e30f;
      #pragma unroll
      for (int j = 0; j < 16; ++j) m = fmaxf(m, Ss[r * 68 + seg * 16 + j]);
      part[tid] = m;
    }
    __syncthreads();
    if (tid < 64) {
      float m = fmaxf(fmaxf(part[tid * 4 + 0], part[tid * 4 + 1]),
                      fmaxf(part[tid * 4 + 2], part[tid * 4 + 3]));
      float mnew = fmaxf(rowm[tid], m);
      rowalpha[tid] = __expf(rowm[tid] - mnew);
      rowm[tid] = mnew;
    }
    __syncthreads();
    { // exp, P (bf16), partial sums
      int r = tid >> 2, seg = tid & 3;
      float mrow = rowm[r], sum = 0.0f;
      #pragma unroll
      for (int j = 0; j < 16; ++j) {
        float e = __expf(Ss[r * 68 + seg * 16 + j] - mrow);
        Ps[r * 72 + seg * 16 + j] = f2bf(e);
        sum += e;
      }
      part[tid] = sum;
    }
    __syncthreads();
    if (tid < 64)
      rowl[tid] = rowl[tid] * rowalpha[tid] +
                  part[tid * 4 + 0] + part[tid * 4 + 1] + part[tid * 4 + 2] + part[tid * 4 + 3];

    // rescale running O by alpha (row dependent)
    #pragma unroll
    for (int v = 0; v < 8; ++v) {
      int r = wm * 16 + v + half * 8;
      float al = rowalpha[r];
      o0[v] *= al; o1[v] *= al;
    }
    // O += P * V
    {
      bf16x16 ap0 = load_frag(Ps, wm * 16 + lc, 72, 0);
      bf16x16 ap1 = load_frag(Ps, wm * 16 + lc, 72, 32);
      bf16x16 bv00 = load_frag(Vs, (wn2 + 0) * 16 + lc, 72, 0);
      bf16x16 bv01 = load_frag(Vs, (wn2 + 0) * 16 + lc, 72, 32);
      bf16x16 bv10 = load_frag(Vs, (wn2 + 1) * 16 + lc, 72, 0);
      bf16x16 bv11 = load_frag(Vs, (wn2 + 1) * 16 + lc, 72, 32);
      o0 = wmma_bf16(ap1, bv01, wmma_bf16(ap0, bv00, o0));
      o1 = wmma_bf16(ap1, bv11, wmma_bf16(ap0, bv10, o1));
    }
    __syncthreads();
  }

  if (tid < 64) rowalpha[tid] = 1.0f / rowl[tid];
  __syncthreads();

  #pragma unroll
  for (int v = 0; v < 8; ++v) {
    int r = wm * 16 + v + half * 8;
    float inv = rowalpha[r];
    int srow = qt * 64 + r;
    size_t rb = ((size_t)b * 1024 + srow) * 1024 + h * 64;
    ctx[rb + (wn2 + 0) * 16 + lc] = f2bf(o0[v] * inv);
    ctx[rb + (wn2 + 1) * 16 + lc] = f2bf(o1[v] * inv);
  }
}

// Generic GEMM: C = act(A@Bt^T + bias). grid = (N/64, M/128). Bt is [N x K] (pre-transposed).
__global__ __launch_bounds__(256) void gemm_kernel(
    const u16* __restrict__ A, int lda, const u16* __restrict__ Bt, int ldb,
    const float* __restrict__ bias, float* __restrict__ Cf, u16* __restrict__ Cb,
    int ldc, int K, int relu)
{
  __shared__ u16 As0[128 * 72];
  __shared__ u16 As1[128 * 72];
  __shared__ u16 Bs0[64 * 72];
  __shared__ u16 Bs1[64 * 72];
  const int row0 = blockIdx.y * 128;
  const int col0 = blockIdx.x * 64;

  floatx8 acc[2][2];
  acc[0][0] = zero8(); acc[0][1] = zero8(); acc[1][0] = zero8(); acc[1][1] = zero8();

  gemm_core_128x64(A + (size_t)row0 * lda, lda, Bt + (size_t)col0 * ldb, ldb, K,
                   As0, As1, Bs0, Bs1, acc);

  const int lane = threadIdx.x & 31, wave = threadIdx.x >> 5;
  const int wm = wave & 3, wn = wave >> 2, half = lane >> 4, lc = lane & 15;
  #pragma unroll
  for (int mt = 0; mt < 2; ++mt)
    #pragma unroll
    for (int nt = 0; nt < 2; ++nt)
      #pragma unroll
      for (int v = 0; v < 8; ++v) {
        int r = row0 + wm * 32 + mt * 16 + v + half * 8;
        int c = col0 + wn * 32 + nt * 16 + lc;
        float val = acc[mt][nt][v] + bias[c];
        if (relu) val = fmaxf(val, 0.0f);
        if (Cf) Cf[(size_t)r * ldc + c] = val;
        if (Cb) Cb[(size_t)r * ldc + c] = f2bf(val);
      }
}

// y = LayerNorm(a + b) ; one block per row of 1024, 256 threads x 4 elems.
__global__ __launch_bounds__(256) void add_ln_kernel(
    const float* __restrict__ a, const float* __restrict__ bsrc,
    const float* __restrict__ gamma, const float* __restrict__ beta,
    float* __restrict__ outf, u16* __restrict__ outb)
{
  __shared__ float red[256];
  const int row = blockIdx.x, tid = threadIdx.x;
  const float* ap = a + (size_t)row * 1024;
  const float* bp = bsrc + (size_t)row * 1024;

  float v[4];
  #pragma unroll
  for (int i = 0; i < 4; ++i) { int c = tid + i * 256; v[i] = ap[c] + bp[c]; }

  red[tid] = v[0] + v[1] + v[2] + v[3];
  __syncthreads();
  for (int off = 128; off > 0; off >>= 1) {
    if (tid < off) red[tid] += red[tid + off];
    __syncthreads();
  }
  float mean = red[0] * (1.0f / 1024.0f);
  __syncthreads();

  float s2 = 0.0f;
  #pragma unroll
  for (int i = 0; i < 4; ++i) { float d = v[i] - mean; s2 += d * d; }
  red[tid] = s2;
  __syncthreads();
  for (int off = 128; off > 0; off >>= 1) {
    if (tid < off) red[tid] += red[tid + off];
    __syncthreads();
  }
  float rstd = rsqrtf(red[0] * (1.0f / 1024.0f) + 1e-5f);

  #pragma unroll
  for (int i = 0; i < 4; ++i) {
    int c = tid + i * 256;
    float y = (v[i] - mean) * rstd * gamma[c] + beta[c];
    outf[(size_t)row * 1024 + c] = y;
    if (outb) outb[(size_t)row * 1024 + c] = f2bf(y);
  }
}

// ---------------- launch ----------------

extern "C" void kernel_launch(void* const* d_in, const int* in_sizes, int n_in,
                              void* d_out, int out_size, void* d_ws, size_t ws_size,
                              hipStream_t stream) {
  (void)in_sizes; (void)n_in; (void)out_size; (void)ws_size;
  const float* x   = (const float*)d_in[0];
  const float* Wq  = (const float*)d_in[1];
  const float* bq  = (const float*)d_in[2];
  const float* Wk  = (const float*)d_in[3];
  const float* bk  = (const float*)d_in[4];
  const float* Wv  = (const float*)d_in[5];
  const float* bv  = (const float*)d_in[6];
  const float* Wo  = (const float*)d_in[7];
  const float* bo  = (const float*)d_in[8];
  const float* g1  = (const float*)d_in[9];
  const float* be1 = (const float*)d_in[10];
  const float* W1  = (const float*)d_in[11];
  const float* b1  = (const float*)d_in[12];
  const float* W2  = (const float*)d_in[13];
  const float* b2  = (const float*)d_in[14];
  const float* g2  = (const float*)d_in[15];
  const float* be2 = (const float*)d_in[16];

  char* ws = (char*)d_ws;
  const size_t MB = 1024 * 1024;
  u16*   xb   = (u16*)(ws + 0);          // 8 MB
  u16*   Wqt  = (u16*)(ws + 8  * MB);    // 2 MB each, transposed [h][d][e] / [n][k]
  u16*   Wkt  = (u16*)(ws + 10 * MB);
  u16*   Wvt  = (u16*)(ws + 12 * MB);
  u16*   Wot  = (u16*)(ws + 14 * MB);
  u16*   W1t  = (u16*)(ws + 16 * MB);
  u16*   W2t  = (u16*)(ws + 18 * MB);
  u16*   Qb   = (u16*)(ws + 20 * MB);    // 8 MB
  u16*   Kb   = (u16*)(ws + 28 * MB);    // 8 MB
  u16*   Vtb  = (u16*)(ws + 36 * MB);    // 8 MB, [h][b][d][s]
  u16*   ctxb = (u16*)(ws + 44 * MB);    // 8 MB
  float* attn = (float*)(ws + 52 * MB);  // 16 MB
  // reuse (lifetimes disjoint):
  float* y1f  = (float*)(ws + 20 * MB);  // over Q/K  (16 MB)
  u16*   y1b  = (u16*)(ws + 36 * MB);    // over V    (8 MB)
  u16*   h1b  = (u16*)(ws + 44 * MB);    // over ctx  (8 MB)
  float* ff2  = (float*)(ws + 52 * MB);  // over attn (16 MB)

  // packs: x plain, weights pack+transpose
  pack_bf16_kernel<<<4096, 256, 0, stream>>>(x, xb, 4 * 1024 * 1024);
  packT_bf16_kernel<<<dim3(64, 1, 16), 256, 0, stream>>>(Wq, Wqt, 64);
  packT_bf16_kernel<<<dim3(64, 1, 16), 256, 0, stream>>>(Wk, Wkt, 64);
  packT_bf16_kernel<<<dim3(64, 1, 16), 256, 0, stream>>>(Wv, Wvt, 64);
  packT_bf16_kernel<<<dim3(1024, 1, 1), 256, 0, stream>>>(Wo, Wot, 1024);
  packT_bf16_kernel<<<dim3(1024, 1, 1), 256, 0, stream>>>(W1, W1t, 1024);
  packT_bf16_kernel<<<dim3(1024, 1, 1), 256, 0, stream>>>(W2, W2t, 1024);

  // QKV projections (V stored transposed)
  qkv_kernel<<<dim3(32, 16, 3), 256, 0, stream>>>(xb, Wqt, Wkt, Wvt, bq, bk, bv, Qb, Kb, Vtb);

  // fused attention -> ctx (concat-head layout [B*S, 1024])
  attn_kernel<<<dim3(16, 16, 4), 256, 0, stream>>>(Qb, Kb, Vtb, ctxb);

  // output projection
  gemm_kernel<<<dim3(16, 32), 256, 0, stream>>>(ctxb, 1024, Wot, 1024, bo, attn, (u16*)nullptr, 1024, 1024, 0);

  // y1 = LN(x + attn)
  add_ln_kernel<<<4096, 256, 0, stream>>>(x, attn, g1, be1, y1f, y1b);

  // FF1 with ReLU (bf16 out), FF2 (f32 out)
  gemm_kernel<<<dim3(16, 32), 256, 0, stream>>>(y1b, 1024, W1t, 1024, b1, (float*)nullptr, h1b, 1024, 1024, 1);
  gemm_kernel<<<dim3(16, 32), 256, 0, stream>>>(h1b, 1024, W2t, 1024, b2, ff2, (u16*)nullptr, 1024, 1024, 0);

  // out = LN(y1 + ff)
  add_ln_kernel<<<4096, 256, 0, stream>>>(y1f, ff2, g2, be2, (float*)d_out, (u16*)nullptr);
}
